// SentenceModel_39943195853199
// MI455X (gfx1250) — matrix operand order repaired and checked by reading, not compiled, not run on hardware
//
#include <hip/hip_runtime.h>
#include <math.h>

// ---------------------------------------------------------------------------
// Problem constants (match reference)
// ---------------------------------------------------------------------------
#define BB     512
#define SS     20
#define TT     32
#define QQ     16
#define EE     128
#define HH     128
#define H3     384          // 3*H
#define NPAIR  190
#define OUTV   32000
#define APAD   160          // label+128 fe, padded 129 -> 160 (5*32)

__device__ __constant__ float kBnInv = 0.99999500003749969f;  // 1/sqrt(1+1e-5)

typedef __attribute__((ext_vector_type(16))) _Float16 v16h;
typedef __attribute__((ext_vector_type(8)))  _Float16 v8h;
typedef __attribute__((ext_vector_type(8)))  float    v8f;
typedef __attribute__((ext_vector_type(4)))  int      v4i;

// Async LDS staging path (gfx1250): probe the builtins so the file compiles on
// both ROCm 7.2 (clang-22) and the amdgpu-toolchain (clang-23).
// Probed prototype (from clang diagnostic): (v4i AS1* gsrc, v4i AS3* lds, Ii, Ii)
#if defined(__has_builtin)
#if __has_builtin(__builtin_amdgcn_global_load_async_to_lds_b128) && \
    __has_builtin(__builtin_amdgcn_s_wait_asynccnt)
#define USE_ASYNC_LDS 1
#endif
#endif
#ifndef USE_ASYNC_LDS
#define USE_ASYNC_LDS 0
#endif

#if USE_ASYNC_LDS
typedef v4i __attribute__((address_space(1)))* as1_v4i_p;
typedef v4i __attribute__((address_space(3)))* as3_v4i_p;
#endif

// ---------------------------------------------------------------------------
// WMMA helpers (gfx1250 v_wmma_f32_16x16x32_f16, layouts per CDNA5 ISA 7.12.2)
// ---------------------------------------------------------------------------
__device__ inline v8f wmma16(v16h a, v16h b, v8f c) {
  return __builtin_amdgcn_wmma_f32_16x16x32_f16(
      /*neg_a=*/false, a, /*neg_b=*/false, b,
      /*c_mod=*/(short)0, c, /*reuse_a=*/false, /*reuse_b=*/false);
}

// A tile 16x32 f16, row-major source with leading dim ld (elements).
// Lane m=lane&15 is row m; kb=(lane>>4)*8; halves [kb,kb+8) and [16+kb,+8).
__device__ inline v16h load_a_tile(const _Float16* base, int ld) {
  const int lane = threadIdx.x & 31;
  const int m  = lane & 15;
  const int kb = (lane >> 4) << 3;
  const _Float16* p = base + (size_t)m * ld + kb;
  v8h lo = *(const v8h*)(p);
  v8h hi = *(const v8h*)(p + 16);
  v16h r;
#pragma unroll
  for (int i = 0; i < 8; ++i) { r[i] = lo[i]; r[i + 8] = hi[i]; }
  return r;
}

// B tile (K=32)x(N=16) sourced from weights stored [N][K] row-major.
// Lane n=lane&15 is output channel n; kb=(lane>>4)*8; halves [kb,kb+8) and [16+kb,+8).
__device__ inline v16h load_b_tile(const _Float16* base, int ld) {
  const int lane = threadIdx.x & 31;
  const int n  = lane & 15;
  const int Kb = (lane >> 4) << 3;
  const _Float16* p = base + (size_t)n * ld + Kb;
  v8h lo = *(const v8h*)(p);
  v8h hi = *(const v8h*)(p + 16);
  v16h r;
#pragma unroll
  for (int i = 0; i < 8; ++i) { r[i] = lo[i]; r[i + 8] = hi[i]; }
  return r;
}

// D/C 16x16 f32 tile -> row-major f32 buffer with leading dim ld.
__device__ inline void store_d_tile_f32(float* base, int ld, v8f d) {
  const int lane = threadIdx.x & 31;
  const int n  = lane & 15;
  const int m0 = (lane >> 4) << 3;
#pragma unroll
  for (int r = 0; r < 8; ++r) base[(size_t)(m0 + r) * ld + n] = d[r];
}

__device__ inline float sigmf(float x) { return 1.f / (1.f + __expf(-x)); }

// ---------------------------------------------------------------------------
// f32 -> f16 weight convert
// ---------------------------------------------------------------------------
__global__ void cvt_f16_kernel(const float* __restrict__ src,
                               _Float16* __restrict__ dst, int n) {
  int i = blockIdx.x * blockDim.x + threadIdx.x;
  if (i < n) dst[i] = (_Float16)src[i];
}

// Split g1_w [256,386] into padded W1p/W2p [256,160] and W3 [256,128] (f16)
__global__ void prep_g1w_kernel(const float* __restrict__ g1w,
                                _Float16* __restrict__ W1p,
                                _Float16* __restrict__ W2p,
                                _Float16* __restrict__ W3) {
  int i = blockIdx.x * blockDim.x + threadIdx.x;
  if (i < 256 * APAD) {
    int n = i / APAD, k = i % APAD;
    W1p[i] = (_Float16)((k < 129) ? g1w[n * 386 + k] : 0.f);
    W2p[i] = (_Float16)((k < 129) ? g1w[n * 386 + 129 + k] : 0.f);
  }
  if (i < 256 * 128) {
    int n = i / 128, k = i % 128;
    W3[i] = (_Float16)g1w[n * 386 + 258 + k];
  }
}

// ---------------------------------------------------------------------------
// GRU kernel: persistent weights in LDS, WMMA per step, fused embedding gather
//   block = 64 threads (2 waves), each wave owns a 16-row tile (32 rows/block)
// ---------------------------------------------------------------------------
#define GRU_W_BYTES    (2 * H3 * EE * 2)                               // 196608
#define GRU_WAVE_BYTES (16*128*2 + 16*256*4 + 16*128*4 + 16*128*4)     // 36864
#define GRU_LDS_BYTES  (GRU_W_BYTES + 2 * GRU_WAVE_BYTES)              // 270336

__global__ __launch_bounds__(64) void gru_kernel(
    const int*   __restrict__ tokens,   // [rows, steps]
    const float* __restrict__ emb,      // [V, 128]
    const float* __restrict__ wih,      // [384, 128] f32
    const float* __restrict__ whh,      // [384, 128] f32
    const float* __restrict__ bih,      // [384]
    const float* __restrict__ bhh,      // [384]
    _Float16*    __restrict__ hout,     // [rows, 128] f16
    int steps) {
  extern __shared__ char smem[];
  _Float16* wih_s = (_Float16*)smem;            // 384*128
  _Float16* whh_s = wih_s + H3 * EE;            // 384*128
  const int wave = threadIdx.x >> 5;
  const int lane = threadIdx.x & 31;
  char*  wbase = smem + GRU_W_BYTES + wave * GRU_WAVE_BYTES;
  _Float16* h_s = (_Float16*)wbase;             // 16*128 f16
  float* gRZ = (float*)(wbase + 16 * 128 * 2);  // 16*256 f32
  float* giN = gRZ + 16 * 256;                  // 16*128 f32
  float* ghN = giN + 16 * 128;                  // 16*128 f32

  // cooperative weight load with f32->f16 convert
  for (int i = threadIdx.x; i < H3 * EE; i += blockDim.x) {
    wih_s[i] = (_Float16)wih[i];
    whh_s[i] = (_Float16)whh[i];
  }
  for (int i = lane; i < 16 * 128; i += 32) h_s[i] = (_Float16)0.f;
  __syncthreads();

  const int row0 = blockIdx.x * 32 + wave * 16;
  const int m    = lane & 15;
  const int kb8  = (lane >> 4) << 3;

  for (int t = 0; t < steps; ++t) {
    // ---- A operand for x_t: gather embedding row, convert f32->f16 ----
    const int tok = tokens[(size_t)(row0 + m) * steps + t];
    const float* xrow = emb + (size_t)tok * EE;
    v16h ax[4];
#pragma unroll
    for (int kk = 0; kk < 4; ++kk) {
      const float* p = xrow + kk * 32 + kb8;
#pragma unroll
      for (int i = 0; i < 8; ++i) {
        ax[kk][i]     = (_Float16)p[i];
        ax[kk][i + 8] = (_Float16)p[16 + i];
      }
    }
    // ---- A operand for h (from LDS, written last iteration by this wave) ----
    v16h ah[4];
#pragma unroll
    for (int kk = 0; kk < 4; ++kk) ah[kk] = load_a_tile(h_s + kk * 32, 128);

    // ---- r,z gates: cols [0,256): accumulate x*Wih^T + h*Whh^T + biases ----
    for (int nt = 0; nt < 16; ++nt) {
      const int n = nt * 16 + (lane & 15);
      const float binit = bih[n] + bhh[n];
      v8f acc;
#pragma unroll
      for (int r = 0; r < 8; ++r) acc[r] = binit;
#pragma unroll
      for (int kk = 0; kk < 4; ++kk)
        acc = wmma16(ax[kk], load_b_tile(wih_s + nt * 16 * EE + kk * 32, EE), acc);
#pragma unroll
      for (int kk = 0; kk < 4; ++kk)
        acc = wmma16(ah[kk], load_b_tile(whh_s + nt * 16 * EE + kk * 32, EE), acc);
      store_d_tile_f32(gRZ + nt * 16, 256, acc);
    }
    // ---- n gate: cols [256,384): keep gi_n and gh_n separate ----
    for (int nt = 0; nt < 8; ++nt) {
      const int n = 256 + nt * 16 + (lane & 15);
      v8f ai, ah2;
#pragma unroll
      for (int r = 0; r < 8; ++r) { ai[r] = bih[n]; ah2[r] = bhh[n]; }
#pragma unroll
      for (int kk = 0; kk < 4; ++kk) {
        ai  = wmma16(ax[kk], load_b_tile(wih_s + (256 + nt * 16) * EE + kk * 32, EE), ai);
        ah2 = wmma16(ah[kk], load_b_tile(whh_s + (256 + nt * 16) * EE + kk * 32, EE), ah2);
      }
      store_d_tile_f32(giN + nt * 16, 128, ai);
      store_d_tile_f32(ghN + nt * 16, 128, ah2);
    }
    // ---- gate math (same-wave LDS: in-order, no barrier needed) ----
    for (int idx = lane; idx < 16 * 128; idx += 32) {
      const int r_ = idx >> 7, c = idx & 127;
      const float rg = sigmf(gRZ[r_ * 256 + c]);
      const float zg = sigmf(gRZ[r_ * 256 + 128 + c]);
      const float ng = tanhf(giN[r_ * 128 + c] + rg * ghN[r_ * 128 + c]);
      const float ho = (float)h_s[r_ * 128 + c];
      h_s[r_ * 128 + c] = (_Float16)((1.f - zg) * ng + zg * ho);
    }
  }
  // ---- final hidden -> global f16 ----
  for (int idx = lane; idx < 16 * 128; idx += 32) {
    const int r_ = idx >> 7, c = idx & 127;
    hout[(size_t)(row0 + r_) * 128 + c] = h_s[r_ * 128 + c];
  }
}

// ---------------------------------------------------------------------------
// Generic WMMA GEMM: Y = epilogue(X[M,K]f16 @ W[N,K]^T f16)
//   mode 0: Yf = raw f32                (bias/gamma/beta unused)
//   mode 1: Yh = f16 relu((acc+b)*s+be) with s = BnInv*gamma
//   mode 2: Yf = f32 (acc+b)*s+be       (no relu)
// block = 128 threads (4 waves); block tile 128x64; wave tile 32x64.
// The 64xK weight slab is staged in LDS (async path on toolchains that have
// global_load_async_to_lds_b128) and feeds WMMA B tiles via ds_load_b128.
// M multiple of 128; N multiple of 64; K multiple of 32 (K*128 B <= LDS).
// ---------------------------------------------------------------------------
__global__ __launch_bounds__(128) void gemm_kernel(
    const _Float16* __restrict__ X, const _Float16* __restrict__ W,
    const float* __restrict__ bias, const float* __restrict__ gamma,
    const float* __restrict__ beta,
    _Float16* __restrict__ Yh, float* __restrict__ Yf,
    int M, int N, int K, int mode) {
  extern __shared__ _Float16 Ws[];  // [64][K]
  const int wave = threadIdx.x >> 5;
  const int lane = threadIdx.x & 31;
  const int m0 = blockIdx.y * 128 + wave * 32;
  const int n0 = blockIdx.x * 64;

  // ---- stage the block's weight slab W[n0 .. n0+63][0 .. K) into LDS ----
  {
    const _Float16* wsrc = W + (size_t)n0 * K;
    const int slab = 64 * K;  // multiple of 1024 for all K used here
#if USE_ASYNC_LDS
    for (int i = threadIdx.x * 8; i < slab; i += 128 * 8) {
      __builtin_amdgcn_global_load_async_to_lds_b128(
          (as1_v4i_p)(wsrc + i), (as3_v4i_p)(Ws + i), 0, 0);
    }
    __builtin_amdgcn_s_wait_asynccnt(0);
#else
    for (int i = threadIdx.x * 8; i < slab; i += 128 * 8)
      *(v8h*)(Ws + i) = *(const v8h*)(wsrc + i);
#endif
    __syncthreads();
  }

  v8f acc[2][4];
#pragma unroll
  for (int mt = 0; mt < 2; ++mt)
#pragma unroll
    for (int i = 0; i < 4; ++i)
#pragma unroll
      for (int r = 0; r < 8; ++r) acc[mt][i][r] = 0.f;

  for (int kk = 0; kk < K; kk += 32) {
    v16h a0 = load_a_tile(X + (size_t)m0 * K + kk, K);
    v16h a1 = load_a_tile(X + (size_t)(m0 + 16) * K + kk, K);
    __builtin_prefetch(X + (size_t)m0 * K + kk + 32, 0, 1);  // global_prefetch_b8
#pragma unroll
    for (int nt = 0; nt < 4; ++nt) {
      v16h b = load_b_tile(Ws + (size_t)(nt * 16) * K + kk, K);
      acc[0][nt] = wmma16(a0, b, acc[0][nt]);
      acc[1][nt] = wmma16(a1, b, acc[1][nt]);
    }
  }

  const int nlane = lane & 15;
  const int mo = (lane >> 4) << 3;
#pragma unroll
  for (int mt = 0; mt < 2; ++mt) {
#pragma unroll
    for (int nt = 0; nt < 4; ++nt) {
      const int n = n0 + nt * 16 + nlane;
      float s = 1.f, c = 0.f;
      if (mode != 0) {
        s = kBnInv * gamma[n];
        c = bias[n] * s + beta[n];
      }
#pragma unroll
      for (int r = 0; r < 8; ++r) {
        const size_t om = (size_t)(m0 + mt * 16 + mo + r) * N + n;
        if (mode == 1) {
          float v = acc[mt][nt][r] * s + c;
          Yh[om] = (_Float16)(v > 0.f ? v : 0.f);
        } else if (mode == 2) {
          Yf[om] = acc[mt][nt][r] * s + c;
        } else {
          Yf[om] = acc[mt][nt][r];
        }
      }
    }
  }
}

// ---------------------------------------------------------------------------
// a = [label | fe | zero-pad] as f16 [B*S, 160]
// ---------------------------------------------------------------------------
__global__ void build_a_kernel(const _Float16* __restrict__ fe,
                               _Float16* __restrict__ a, int rows) {
  int i = blockIdx.x * blockDim.x + threadIdx.x;
  if (i >= rows * APAD) return;
  int row = i / APAD, k = i % APAD;
  _Float16 v;
  if (k == 0)        v = (_Float16)(float)((row % SS) + 1);
  else if (k < 129)  v = fe[(size_t)row * 128 + (k - 1)];
  else               v = (_Float16)0.f;
  a[i] = v;
}

// ---------------------------------------------------------------------------
// g1 assemble: out[b,p,n] = relu(bn(P[b,i]+Q[b,j]+R[b]+b1)) -> f16 [B*190,256]
// ---------------------------------------------------------------------------
__global__ void assemble_g1_kernel(const float* __restrict__ P,
                                   const float* __restrict__ Qm,
                                   const float* __restrict__ R,
                                   const float* __restrict__ b1,
                                   const float* __restrict__ g1,
                                   const float* __restrict__ be1,
                                   _Float16* __restrict__ out) {
  long long i = (long long)blockIdx.x * blockDim.x + threadIdx.x;
  const long long total = (long long)BB * NPAIR * 256;
  if (i >= total) return;
  const int n = (int)(i & 255);
  long long t = i >> 8;
  const int p = (int)(t % NPAIR);
  const int b = (int)(t / NPAIR);
  // decode pair index -> (ii, jj) in combinations(range(20),2) order
  int ii = 0, rem = p, span = SS - 1;
  while (rem >= span) { rem -= span; --span; ++ii; }
  const int jj = ii + 1 + rem;
  float v = P[(size_t)(b * SS + ii) * 256 + n] +
            Qm[(size_t)(b * SS + jj) * 256 + n] +
            R[(size_t)b * 256 + n] + b1[n];
  const float s = kBnInv * g1[n];
  v = v * s + be1[n];
  out[i] = (_Float16)(v > 0.f ? v : 0.f);
}

// ---------------------------------------------------------------------------
// Sum over the 190 pairs -> g_sum f16 [B,256]
// ---------------------------------------------------------------------------
__global__ void pair_sum_kernel(const _Float16* __restrict__ g4,
                                _Float16* __restrict__ gsum) {
  const int b = blockIdx.x;
  const int n = threadIdx.x;  // 256 threads
  float s = 0.f;
  for (int p = 0; p < NPAIR; ++p)
    s += (float)g4[((size_t)b * NPAIR + p) * 256 + n];
  gsum[(size_t)b * 256 + n] = (_Float16)s;
}

// ---------------------------------------------------------------------------
// Row-wise log-softmax over 32000
// ---------------------------------------------------------------------------
__global__ __launch_bounds__(256) void logsoftmax_kernel(
    const float* __restrict__ X, float* __restrict__ Y, int N) {
  __shared__ float red[256];
  const int b = blockIdx.x;
  const float* xr = X + (size_t)b * N;
  float mx = -3.0e38f;
  for (int i = threadIdx.x; i < N; i += 256) mx = fmaxf(mx, xr[i]);
  red[threadIdx.x] = mx; __syncthreads();
  for (int s = 128; s > 0; s >>= 1) {
    if (threadIdx.x < s) red[threadIdx.x] = fmaxf(red[threadIdx.x], red[threadIdx.x + s]);
    __syncthreads();
  }
  mx = red[0]; __syncthreads();
  float sm = 0.f;
  for (int i = threadIdx.x; i < N; i += 256) sm += __expf(xr[i] - mx);
  red[threadIdx.x] = sm; __syncthreads();
  for (int s = 128; s > 0; s >>= 1) {
    if (threadIdx.x < s) red[threadIdx.x] += red[threadIdx.x + s];
    __syncthreads();
  }
  const float lse = mx + __logf(red[0]);
  float* yr = Y + (size_t)b * N;
  for (int i = threadIdx.x; i < N; i += 256) yr[i] = xr[i] - lse;
}

// ---------------------------------------------------------------------------
// Host launcher
// ---------------------------------------------------------------------------
extern "C" void kernel_launch(void* const* d_in, const int* in_sizes, int n_in,
                              void* d_out, int out_size, void* d_ws, size_t ws_size,
                              hipStream_t stream) {
  (void)in_sizes; (void)n_in; (void)out_size; (void)ws_size;
  // ---- inputs (setup_inputs order) ----
  const int*   story     = (const int*)d_in[0];     // [B,S,T]
  const int*   query     = (const int*)d_in[1];     // [B,Q]
  const float* story_emb = (const float*)d_in[6];
  const float* query_emb = (const float*)d_in[7];
  const float* q_wih  = (const float*)d_in[8];
  const float* q_whh  = (const float*)d_in[9];
  const float* q_bih  = (const float*)d_in[10];
  const float* q_bhh  = (const float*)d_in[11];
  const float* fr_wih = (const float*)d_in[12];
  const float* fr_whh = (const float*)d_in[13];
  const float* fr_bih = (const float*)d_in[14];
  const float* fr_bhh = (const float*)d_in[15];
  const float* g1_w = (const float*)d_in[16];
  const float* g1_b = (const float*)d_in[17];
  const float* g1_g = (const float*)d_in[18];
  const float* g1_be= (const float*)d_in[19];
  const float* g2_w = (const float*)d_in[20];
  const float* g2_b = (const float*)d_in[21];
  const float* g2_g = (const float*)d_in[22];
  const float* g2_be= (const float*)d_in[23];
  const float* g3_w = (const float*)d_in[24];
  const float* g3_b = (const float*)d_in[25];
  const float* g3_g = (const float*)d_in[26];
  const float* g3_be= (const float*)d_in[27];
  const float* g4_w = (const float*)d_in[28];
  const float* g4_b = (const float*)d_in[29];
  const float* g4_g = (const float*)d_in[30];
  const float* g4_be= (const float*)d_in[31];
  const float* f1_w = (const float*)d_in[32];
  const float* f1_b = (const float*)d_in[33];
  const float* f1_g = (const float*)d_in[34];
  const float* f1_be= (const float*)d_in[35];
  const float* f2_w = (const float*)d_in[36];
  const float* f2_b = (const float*)d_in[37];
  const float* f2_g = (const float*)d_in[38];
  const float* f2_be= (const float*)d_in[39];
  const float* f3_w = (const float*)d_in[40];
  const float* f3_b = (const float*)d_in[41];
  const float* f3_g = (const float*)d_in[42];
  const float* f3_be= (const float*)d_in[43];
  float* out = (float*)d_out;

  // ---- workspace bump allocator ----
  char* base = (char*)d_ws;
  size_t off = 0;
  auto alloc = [&](size_t bytes) -> void* {
    void* p = base + off;
    off = (off + bytes + 255) & ~(size_t)255;
    return p;
  };
  _Float16* fe16   = (_Float16*)alloc((size_t)BB * SS * 128 * 2);
  _Float16* qc16   = (_Float16*)alloc((size_t)BB * 128 * 2);
  _Float16* a16    = (_Float16*)alloc((size_t)BB * SS * APAD * 2);
  _Float16* W1p    = (_Float16*)alloc((size_t)256 * APAD * 2);
  _Float16* W2p    = (_Float16*)alloc((size_t)256 * APAD * 2);
  _Float16* W3     = (_Float16*)alloc((size_t)256 * 128 * 2);
  _Float16* wg2    = (_Float16*)alloc((size_t)256 * 256 * 2);
  _Float16* wg3    = (_Float16*)alloc((size_t)256 * 256 * 2);
  _Float16* wg4    = (_Float16*)alloc((size_t)256 * 256 * 2);
  _Float16* wf1    = (_Float16*)alloc((size_t)256 * 256 * 2);
  _Float16* wf2    = (_Float16*)alloc((size_t)512 * 256 * 2);
  _Float16* wf3    = (_Float16*)alloc((size_t)OUTV * 512 * 2);
  float*    P32    = (float*)alloc((size_t)BB * SS * 256 * 4);
  float*    Q32    = (float*)alloc((size_t)BB * SS * 256 * 4);
  float*    R32    = (float*)alloc((size_t)BB * 256 * 4);
  _Float16* gA     = (_Float16*)alloc((size_t)BB * NPAIR * 256 * 2);
  _Float16* gB     = (_Float16*)alloc((size_t)BB * NPAIR * 256 * 2);
  _Float16* gsum16 = (_Float16*)alloc((size_t)BB * 256 * 2);
  _Float16* fb1    = (_Float16*)alloc((size_t)BB * 256 * 2);
  _Float16* fb2    = (_Float16*)alloc((size_t)BB * 512 * 2);
  float*    f3out  = (float*)alloc((size_t)BB * OUTV * 4);

  auto cvt = [&](const float* s, _Float16* d, int n) {
    cvt_f16_kernel<<<(n + 255) / 256, 256, 0, stream>>>(s, d, n);
  };
  // ---- weight conversions ----
  cvt(g2_w, wg2, 256 * 256);
  cvt(g3_w, wg3, 256 * 256);
  cvt(g4_w, wg4, 256 * 256);
  cvt(f1_w, wf1, 256 * 256);
  cvt(f2_w, wf2, 512 * 256);
  cvt(f3_w, wf3, OUTV * 512);
  prep_g1w_kernel<<<(256 * APAD + 255) / 256, 256, 0, stream>>>(g1_w, W1p, W2p, W3);

  // ---- GRUs (WMMA recurrence; weights resident in LDS) ----
  gru_kernel<<<(BB * SS) / 32, 64, GRU_LDS_BYTES, stream>>>(
      story, story_emb, fr_wih, fr_whh, fr_bih, fr_bhh, fe16, TT);
  gru_kernel<<<BB / 32, 64, GRU_LDS_BYTES, stream>>>(
      query, query_emb, q_wih, q_whh, q_bih, q_bhh, qc16, QQ);

  // generic GEMM launcher: block tile 128x64, LDS = 64*K*2 bytes
  auto gemm = [&](const _Float16* X, const _Float16* W,
                  const float* b, const float* g, const float* be,
                  _Float16* Yh, float* Yf, int M, int N, int K, int mode) {
    gemm_kernel<<<dim3(N / 64, M / 128), 128, (size_t)64 * K * 2, stream>>>(
        X, W, b, g, be, Yh, Yf, M, N, K, mode);
  };

  // ---- pair features via g1 decomposition ----
  build_a_kernel<<<(BB * SS * APAD + 255) / 256, 256, 0, stream>>>(fe16, a16, BB * SS);
  gemm(a16, W1p, nullptr, nullptr, nullptr, nullptr, P32, BB * SS, 256, APAD, 0);
  gemm(a16, W2p, nullptr, nullptr, nullptr, nullptr, Q32, BB * SS, 256, APAD, 0);
  gemm(qc16, W3, nullptr, nullptr, nullptr, nullptr, R32, BB, 256, 128, 0);
  {
    long long total = (long long)BB * NPAIR * 256;
    assemble_g1_kernel<<<(unsigned)((total + 255) / 256), 256, 0, stream>>>(
        P32, Q32, R32, g1_b, g1_g, g1_be, gA);
  }

  // ---- g2..g4 (fused GEMM + BN + ReLU) ----
  const int MP = BB * NPAIR;  // 97280
  gemm(gA, wg2, g2_b, g2_g, g2_be, gB, nullptr, MP, 256, 256, 1);
  gemm(gB, wg3, g3_b, g3_g, g3_be, gA, nullptr, MP, 256, 256, 1);
  gemm(gA, wg4, g4_b, g4_g, g4_be, gB, nullptr, MP, 256, 256, 1);

  // ---- pair sum + f-MLP ----
  pair_sum_kernel<<<BB, 256, 0, stream>>>(gB, gsum16);
  gemm(gsum16, wf1, f1_b, f1_g, f1_be, fb1, nullptr, BB, 256, 256, 1);
  gemm(fb1, wf2, f2_b, f2_g, f2_be, fb2, nullptr, BB, 512, 256, 1);
  gemm(fb2, wf3, f3_b, f3_g, f3_be, nullptr, f3out, BB, OUTV, 512, 2);

  // ---- log-softmax -> output ----
  logsoftmax_kernel<<<BB, 256, 0, stream>>>(f3out, out, OUTV);
}
